// CausalSelfAttention_10960756539678
// MI455X (gfx1250) — matrix operand order repaired and checked by reading, not compiled, not run on hardware
//
#include <hip/hip_runtime.h>
#include <math.h>

#define DTOT   1024
#define HEADS  16
#define HD     64
#define BB     4
#define TT     2048
#define M_ROWS (BB*TT)     // 8192
#define N_QKV  (3*DTOT)    // 3072

typedef __attribute__((ext_vector_type(16))) __bf16 v16bf;
typedef __attribute__((ext_vector_type(8)))  __bf16 v8bf;
typedef __attribute__((ext_vector_type(8)))  float  v8f;

static __device__ inline v8f wmma_bf16(v16bf a, v16bf b, v8f c) {
  // D = A(16x32 bf16) * B(32x16 bf16) + C(16x16 f32)
  return __builtin_amdgcn_wmma_f32_16x16x32_bf16(false, a, false, b, (short)0, c,
                                                 false, false);
}

// A-fragment (16-bit 16x32): lane holds row m=lane%16; K elements are two
// contiguous runs of 8: [half*8, half*8+8) and [16+half*8, 16+half*8+8).
static __device__ inline v16bf load_afrag(const __bf16* rowptr, int half) {
  v8bf lo = *(const v8bf*)(rowptr + half * 8);
  v8bf hi = *(const v8bf*)(rowptr + 16 + half * 8);
  v16bf a;
#pragma unroll
  for (int i = 0; i < 8; ++i) { a[i] = lo[i]; a[8 + i] = hi[i]; }
  return a;
}

static __device__ inline v16bf load_bfrag(const __bf16* colptr, int half, int ld) {
  v16bf b;
#pragma unroll
  for (int i = 0; i < 8; ++i) {
    b[i] = colptr[(size_t)(half * 8 + i) * ld];
    b[8 + i] = colptr[(size_t)(16 + half * 8 + i) * ld];
  }
  return b;
}

__global__ void cvt_f32_bf16(const float* __restrict__ src,
                             __bf16* __restrict__ dst, int n) {
  int i = blockIdx.x * blockDim.x + threadIdx.x;
  int stride = gridDim.x * blockDim.x;
  for (; i < n; i += stride) dst[i] = (__bf16)src[i];
}

// ---------------------------------------------------------------------------
// Software-pipelined WMMA GEMM core: wave computes a 32x64 tile over K=DTOT.
// Fragments for stage n+1 are fetched before the stage-n WMMAs issue, so the
// loadcnt waits retire against loads issued a full WMMA-stage earlier.
// ---------------------------------------------------------------------------
template <int NN>
static __device__ inline void gemm_core(const __bf16* __restrict__ arow0,
                                        const __bf16* __restrict__ arow1,
                                        const __bf16* __restrict__ brow0,
                                        int half, v8f acc[2][4]) {
  v16bf a0c = load_afrag(arow0, half);
  v16bf a1c = load_afrag(arow1, half);
  v16bf bc[4];
#pragma unroll
  for (int ns = 0; ns < 4; ++ns) bc[ns] = load_bfrag(brow0 + ns * 16, half, NN);

  for (int kk = 0; kk < DTOT; kk += 64) {
    // Prefetch stage n+1 (kk+32) before issuing stage-n WMMAs.
    const __bf16* brn = brow0 + (size_t)(kk + 32) * NN;
    v16bf a0n = load_afrag(arow0 + kk + 32, half);
    v16bf a1n = load_afrag(arow1 + kk + 32, half);
    v16bf bn[4];
#pragma unroll
    for (int ns = 0; ns < 4; ++ns) bn[ns] = load_bfrag(brn + ns * 16, half, NN);

#pragma unroll
    for (int ns = 0; ns < 4; ++ns) {
      acc[0][ns] = wmma_bf16(a0c, bc[ns], acc[0][ns]);
      acc[1][ns] = wmma_bf16(a1c, bc[ns], acc[1][ns]);
    }

    if (kk + 64 < DTOT) {  // prefetch stage n+2 (kk+64)
      const __bf16* brc = brow0 + (size_t)(kk + 64) * NN;
      a0c = load_afrag(arow0 + kk + 64, half);
      a1c = load_afrag(arow1 + kk + 64, half);
#pragma unroll
      for (int ns = 0; ns < 4; ++ns) bc[ns] = load_bfrag(brc + ns * 16, half, NN);
    }

#pragma unroll
    for (int ns = 0; ns < 4; ++ns) {
      acc[0][ns] = wmma_bf16(a0n, bn[ns], acc[0][ns]);
      acc[1][ns] = wmma_bf16(a1n, bn[ns], acc[1][ns]);
    }
  }
}

// ---------------------------------------------------------------------------
// QKV GEMM: C[8192,3072] = xb @ Wb + bias ; scatter into Q (scaled), Kt, V
// 8 waves/block; each wave computes a 32x64 tile. Block tile 128x128.
// ---------------------------------------------------------------------------
__global__ __launch_bounds__(256) void gemm_qkv(
    const __bf16* __restrict__ xb, const __bf16* __restrict__ wb,
    const float* __restrict__ bias, __bf16* __restrict__ Qb,
    __bf16* __restrict__ Ktb, __bf16* __restrict__ Vb) {
  const int lane = threadIdx.x & 31, half = lane >> 4, lr = lane & 15;
  const int wid = threadIdx.x >> 5;
  const int nblk = N_QKV / 128;  // 24
  const int bm = blockIdx.x / nblk, bn = blockIdx.x % nblk;
  const int m_base = bm * 128 + (wid & 3) * 32;
  const int n_base = bn * 128 + (wid >> 2) * 64;

  v8f acc[2][4];
#pragma unroll
  for (int ms = 0; ms < 2; ++ms)
#pragma unroll
    for (int ns = 0; ns < 4; ++ns)
#pragma unroll
      for (int j = 0; j < 8; ++j) acc[ms][ns][j] = 0.0f;

  const __bf16* arow0 = xb + (size_t)(m_base + lr) * DTOT;
  const __bf16* arow1 = xb + (size_t)(m_base + 16 + lr) * DTOT;
  const __bf16* brow0 = wb + lr + n_base;
  gemm_core<N_QKV>(arow0, arow1, brow0, half, acc);

#pragma unroll
  for (int ns = 0; ns < 4; ++ns) {
    const int c = n_base + ns * 16 + lr;
    const float bv = bias[c];
    const int sec = c >> 10;          // 0=q 1=k 2=v
    const int hc = c & 1023;
    const int head = hc >> 6, d = hc & 63;
#pragma unroll
    for (int ms = 0; ms < 2; ++ms) {
#pragma unroll
      for (int j = 0; j < 8; ++j) {
        const int r = m_base + ms * 16 + j + 8 * half;
        const int bidx = r >> 11, t = r & (TT - 1);
        const float v = acc[ms][ns][j] + bv;
        const size_t hb = (size_t)bidx * HEADS + head;
        if (sec == 0)
          Qb[(hb * TT + t) * HD + d] = (__bf16)(v * 0.125f);  // 1/sqrt(64)
        else if (sec == 1)
          Ktb[(hb * HD + d) * TT + t] = (__bf16)v;            // transposed K
        else
          Vb[(hb * TT + t) * HD + d] = (__bf16)v;
      }
    }
  }
}

// ---------------------------------------------------------------------------
// Flash attention: one wave per 16-query tile; 32 keys per chunk.
// ---------------------------------------------------------------------------
__global__ __launch_bounds__(256) void attn(const __bf16* __restrict__ Qb,
                                            const __bf16* __restrict__ Ktb,
                                            const __bf16* __restrict__ Vb,
                                            __bf16* __restrict__ Yb) {
  __shared__ __bf16 ldsP[8][16 * 32];  // per-wave 16x32 P staging (8 KB)
  const int lane = threadIdx.x & 31, half = lane >> 4, lr = lane & 15;
  const int wid = threadIdx.x >> 5;
  const int w = blockIdx.x * 8 + wid;
  const int qt = w & 127;
  const int h = (w >> 7) & 15;
  const int b = w >> 11;
  const int q0 = qt * 16;
  const size_t bh = (size_t)b * HEADS + h;

  const __bf16* qrow = Qb + (bh * TT + q0 + lr) * HD;
  const v16bf aQ0 = load_afrag(qrow, half);        // d = 0..31
  const v16bf aQ1 = load_afrag(qrow + 32, half);   // d = 32..63

  v8f acc[4];
  float mrow[8], lrow[8];
#pragma unroll
  for (int nt = 0; nt < 4; ++nt)
#pragma unroll
    for (int j = 0; j < 8; ++j) acc[nt][j] = 0.0f;
#pragma unroll
  for (int j = 0; j < 8; ++j) { mrow[j] = -__builtin_inff(); lrow[j] = 0.0f; }

  __bf16* myP = &ldsP[wid][0];
  const __bf16* ktbase = Ktb + bh * HD * TT;
  const __bf16* vbase = Vb + bh * TT * HD;
  const int brow = lr + 16 * half;  // B-fragment row within 32-chunk

  for (int k0 = 0; k0 <= q0 + 15; k0 += 32) {
    v8f s0, s1;
#pragma unroll
    for (int j = 0; j < 8; ++j) { s0[j] = 0.0f; s1[j] = 0.0f; }
    {
      v16bf bk00 = load_bfrag(ktbase + k0 + lr, half, TT);
      v16bf bk01 = load_bfrag(ktbase + k0 + 16 + lr, half, TT);
      v16bf bk10 = load_bfrag(ktbase + (size_t)32 * TT + k0 + lr, half, TT);
      v16bf bk11 = load_bfrag(ktbase + (size_t)32 * TT + k0 + 16 + lr, half, TT);
      s0 = wmma_bf16(aQ0, bk00, s0);
      s0 = wmma_bf16(aQ1, bk10, s0);
      s1 = wmma_bf16(aQ0, bk01, s1);
      s1 = wmma_bf16(aQ1, bk11, s1);
    }
    if (k0 + 31 > q0) {  // diagonal chunk: causal mask per element
#pragma unroll
      for (int j = 0; j < 8; ++j) {
        const int q = q0 + j + 8 * half;
        if (k0 + lr > q)      s0[j] = -__builtin_inff();
        if (k0 + 16 + lr > q) s1[j] = -__builtin_inff();
      }
    }
    // Online softmax: per C-row stats live in (VGPR j, lane-half) pairs.
#pragma unroll
    for (int j = 0; j < 8; ++j) {
      float cm = fmaxf(s0[j], s1[j]);
      cm = fmaxf(cm, __shfl_xor(cm, 1));
      cm = fmaxf(cm, __shfl_xor(cm, 2));
      cm = fmaxf(cm, __shfl_xor(cm, 4));
      cm = fmaxf(cm, __shfl_xor(cm, 8));
      const float nm = fmaxf(mrow[j], cm);
      const float scale = __expf(mrow[j] - nm);
      const float p0 = __expf(s0[j] - nm);
      const float p1 = __expf(s1[j] - nm);
      float rs = p0 + p1;
      rs += __shfl_xor(rs, 1);
      rs += __shfl_xor(rs, 2);
      rs += __shfl_xor(rs, 4);
      rs += __shfl_xor(rs, 8);
      lrow[j] = lrow[j] * scale + rs;
      mrow[j] = nm;
#pragma unroll
      for (int nt = 0; nt < 4; ++nt) acc[nt][j] = acc[nt][j] * scale;
      const int m = j + 8 * half;
      myP[m * 32 + lr] = (__bf16)p0;
      myP[m * 32 + 16 + lr] = (__bf16)p1;
    }
    asm volatile("s_wait_dscnt 0" ::: "memory");  // C-layout -> A-layout via LDS
    const v16bf aP = load_afrag(myP + (size_t)lr * 32, half);
#pragma unroll
    for (int nt = 0; nt < 4; ++nt) {
      v16bf bv = load_bfrag(vbase + (size_t)k0 * HD + nt * 16 + lr, half, HD);
      acc[nt] = wmma_bf16(aP, bv, acc[nt]);
    }
  }

#pragma unroll
  for (int j = 0; j < 8; ++j) {
    const float inv = 1.0f / lrow[j];
    const int q = q0 + j + 8 * half;
    const size_t orow = ((size_t)b * TT + q) * DTOT + (size_t)h * HD;
#pragma unroll
    for (int nt = 0; nt < 4; ++nt)
      Yb[orow + nt * 16 + lr] = (__bf16)(acc[nt][j] * inv);
  }
}

// ---------------------------------------------------------------------------
// Projection GEMM: out[8192,1024] = Yb @ Wpb + b_proj (f32 output)
// ---------------------------------------------------------------------------
__global__ __launch_bounds__(256) void gemm_proj(
    const __bf16* __restrict__ yb, const __bf16* __restrict__ wb,
    const float* __restrict__ bias, float* __restrict__ out) {
  const int lane = threadIdx.x & 31, half = lane >> 4, lr = lane & 15;
  const int wid = threadIdx.x >> 5;
  const int nblk = DTOT / 128;  // 8
  const int bm = blockIdx.x / nblk, bn = blockIdx.x % nblk;
  const int m_base = bm * 128 + (wid & 3) * 32;
  const int n_base = bn * 128 + (wid >> 2) * 64;

  v8f acc[2][4];
#pragma unroll
  for (int ms = 0; ms < 2; ++ms)
#pragma unroll
    for (int ns = 0; ns < 4; ++ns)
#pragma unroll
      for (int j = 0; j < 8; ++j) acc[ms][ns][j] = 0.0f;

  const __bf16* arow0 = yb + (size_t)(m_base + lr) * DTOT;
  const __bf16* arow1 = yb + (size_t)(m_base + 16 + lr) * DTOT;
  const __bf16* brow0 = wb + lr + n_base;
  gemm_core<DTOT>(arow0, arow1, brow0, half, acc);

#pragma unroll
  for (int ns = 0; ns < 4; ++ns) {
    const int c = n_base + ns * 16 + lr;
    const float bv = bias[c];
#pragma unroll
    for (int ms = 0; ms < 2; ++ms)
#pragma unroll
      for (int j = 0; j < 8; ++j) {
        const int r = m_base + ms * 16 + j + 8 * half;
        out[(size_t)r * DTOT + c] = acc[ms][ns][j] + bv;
      }
  }
}

// ---------------------------------------------------------------------------
extern "C" void kernel_launch(void* const* d_in, const int* in_sizes, int n_in,
                              void* d_out, int out_size, void* d_ws,
                              size_t ws_size, hipStream_t stream) {
  const float* x = (const float*)d_in[0];
  const float* W_attn = (const float*)d_in[1];
  const float* b_attn = (const float*)d_in[2];
  const float* W_proj = (const float*)d_in[3];
  const float* b_proj = (const float*)d_in[4];
  float* out = (float*)d_out;

  char* ws = (char*)d_ws;
  size_t off = 0;
  auto wsalloc = [&](size_t bytes) -> void* {
    void* p = ws + off;
    off += (bytes + 255) & ~(size_t)255;
    return p;
  };
  __bf16* xb  = (__bf16*)wsalloc((size_t)M_ROWS * DTOT * 2);
  __bf16* Wab = (__bf16*)wsalloc((size_t)DTOT * N_QKV * 2);
  __bf16* Wpb = (__bf16*)wsalloc((size_t)DTOT * DTOT * 2);
  __bf16* Qb  = (__bf16*)wsalloc((size_t)BB * HEADS * TT * HD * 2);
  __bf16* Ktb = (__bf16*)wsalloc((size_t)BB * HEADS * TT * HD * 2);
  __bf16* Vb  = (__bf16*)wsalloc((size_t)BB * HEADS * TT * HD * 2);
  __bf16* Yb  = (__bf16*)wsalloc((size_t)M_ROWS * DTOT * 2);

  const int nx = M_ROWS * DTOT;
  const int nwa = DTOT * N_QKV;
  const int nwp = DTOT * DTOT;
  cvt_f32_bf16<<<4096, 256, 0, stream>>>(x, xb, nx);
  cvt_f32_bf16<<<4096, 256, 0, stream>>>(W_attn, Wab, nwa);
  cvt_f32_bf16<<<2048, 256, 0, stream>>>(W_proj, Wpb, nwp);

  gemm_qkv<<<(M_ROWS / 128) * (N_QKV / 128), 256, 0, stream>>>(
      xb, Wab, b_attn, Qb, Ktb, Vb);

  attn<<<BB * HEADS * (TT / 16) / 8, 256, 0, stream>>>(Qb, Ktb, Vb, Yb);

  gemm_proj<<<(M_ROWS / 128) * (DTOT / 128), 256, 0, stream>>>(Yb, Wpb, b_proj,
                                                               out);
}
